// SAGE_mgaev2_266287972767
// MI455X (gfx1250) — matrix operand order repaired and checked by reading, not compiled, not run on hardware
//
#include <hip/hip_runtime.h>
#include <hip/hip_bf16.h>

// ---------------------------------------------------------------------------
// Types for CDNA5 WMMA (wave32)
// ---------------------------------------------------------------------------
typedef __attribute__((ext_vector_type(16))) __bf16 v16bf;
typedef __attribute__((ext_vector_type(8)))  float  v8f;
typedef unsigned int u32;
typedef __attribute__((ext_vector_type(4))) u32 u32x4;

struct U32x8 { u32x4 lo; u32x4 hi; };

__device__ __forceinline__ unsigned short f32_to_bf16(float f) {
    // round-to-nearest-even truncation to bf16
    u32 u = __builtin_bit_cast(u32, f);
    u32 r = u + 0x7FFFu + ((u >> 16) & 1u);
    return (unsigned short)(r >> 16);
}

// Build a v16bf fragment from two 16-byte chunks (LDS or global).
__device__ __forceinline__ v16bf frag2(const unsigned short* p0,
                                       const unsigned short* p1) {
    U32x8 s;
    s.lo = *(const u32x4*)p0;
    s.hi = *(const u32x4*)p1;
    return __builtin_bit_cast(v16bf, s);
}

// ---------------------------------------------------------------------------
// Utility kernels
// ---------------------------------------------------------------------------
__global__ void k_zero_f32(float* __restrict__ p, int n) {
    int i = blockIdx.x * blockDim.x + threadIdx.x;
    if (i < n) p[i] = 0.0f;
}

__global__ void k_degree(const int* __restrict__ row, float* __restrict__ deg, int E) {
    int e = blockIdx.x * blockDim.x + threadIdx.x;
    if (e < E) atomicAdd(&deg[row[e]], 1.0f);
}

// Convert 128x128 f32 weight W[k][n] -> bf16 transposed WT[n][k]
__global__ void k_wconv(const float* __restrict__ W, unsigned short* __restrict__ WT) {
    int i = blockIdx.x * blockDim.x + threadIdx.x;   // 16384 threads
    int n = i >> 7, k = i & 127;
    WT[n * 128 + k] = f32_to_bf16(W[k * 128 + n]);
}

// ---------------------------------------------------------------------------
// Edge scatter: one wave (32 lanes) per edge, each lane moves 4 floats.
// agg[row[e], :] += h[col[e], :]
// ---------------------------------------------------------------------------
__global__ void k_scatter(const float* __restrict__ h,
                          const int* __restrict__ row,
                          const int* __restrict__ col,
                          float* __restrict__ agg, int E) {
    int tid = blockIdx.x * blockDim.x + threadIdx.x;
    int e = tid >> 5;
    if (e >= E) return;
    int q = tid & 31;
    int src = col[e];
    int dst = row[e];
    const float4 v = *(const float4*)(h + (size_t)src * 128 + q * 4);
    float* o = agg + (size_t)dst * 128 + q * 4;
    atomicAdd(o + 0, v.x);
    atomicAdd(o + 1, v.y);
    atomicAdd(o + 2, v.z);
    atomicAdd(o + 3, v.w);
}

// ---------------------------------------------------------------------------
// Fused SAGE dense step:  out = relu( (agg/deg) @ Wl + h @ Wr + b )
// 64 rows per block, 8 waves; each wave computes a 16x64 output slab as
// 4 x (16x16) f32 accumulators via v_wmma_f32_16x16x32_bf16.
// WlT/WrT are bf16, transposed to [n][k] so B-fragments are contiguous.
// ---------------------------------------------------------------------------
__global__ void __launch_bounds__(256)
k_sage_gemm(const float* __restrict__ agg,
            const float* __restrict__ deg,
            const float* __restrict__ hin,
            const unsigned short* __restrict__ WlT,
            const unsigned short* __restrict__ WrT,
            const float* __restrict__ bias,
            float* __restrict__ hout, int N) {
    __shared__ __attribute__((aligned(16))) unsigned short sAgg[64 * 128];
    __shared__ __attribute__((aligned(16))) unsigned short sH[64 * 128];

    const int t = threadIdx.x;
    const int rowBase = blockIdx.x * 64;

    // Stage activation tiles to LDS, converting f32 -> bf16 (mean-normalize agg).
    for (int i = t; i < 64 * 128; i += 256) {
        int r = i >> 7, c = i & 127;
        int g = rowBase + r;
        float va = 0.0f, vh = 0.0f;
        if (g < N) {
            float rd = 1.0f / fmaxf(deg[g], 1.0f);
            va = agg[(size_t)g * 128 + c] * rd;
            vh = hin[(size_t)g * 128 + c];
        }
        sAgg[i] = f32_to_bf16(va);
        sH[i]   = f32_to_bf16(vh);
    }
    __syncthreads();

    const int lane  = t & 31;
    const int wave  = t >> 5;
    const int m0    = (wave >> 1) * 16;        // M tile of this wave (0..48)
    const int n0    = (wave & 1) * 64;         // N half of this wave (0 or 64)
    const int mrow  = m0 + (lane & 15);        // A-fragment row for this lane
    const int khalf = lane >> 4;               // K-half select per ISA layout
    const int ncol  = lane & 15;

    v8f acc[4] = {v8f{}, v8f{}, v8f{}, v8f{}};

    // agg_norm @ Wl
    #pragma unroll
    for (int kb = 0; kb < 128; kb += 32) {
        const unsigned short* ap = &sAgg[mrow * 128 + kb + khalf * 8];
        v16bf a = frag2(ap, ap + 16);
        #pragma unroll
        for (int j = 0; j < 4; ++j) {
            const unsigned short* bp = WlT + (n0 + j * 16 + ncol) * 128 + kb + khalf * 8;
            v16bf b = frag2(bp, bp + 16);
            acc[j] = __builtin_amdgcn_wmma_f32_16x16x32_bf16(
                false, a, false, b, (short)0, acc[j], false, false);
        }
    }
    // + h @ Wr  (same accumulators)
    #pragma unroll
    for (int kb = 0; kb < 128; kb += 32) {
        const unsigned short* ap = &sH[mrow * 128 + kb + khalf * 8];
        v16bf a = frag2(ap, ap + 16);
        #pragma unroll
        for (int j = 0; j < 4; ++j) {
            const unsigned short* bp = WrT + (n0 + j * 16 + ncol) * 128 + kb + khalf * 8;
            v16bf b = frag2(bp, bp + 16);
            acc[j] = __builtin_amdgcn_wmma_f32_16x16x32_bf16(
                false, a, false, b, (short)0, acc[j], false, false);
        }
    }

    // D layout (16x16 f32): VGPR r -> row m0 + (lane>>4)*8 + r, col = lane&15.
    const int mBase = rowBase + m0 + khalf * 8;
    #pragma unroll
    for (int j = 0; j < 4; ++j) {
        const int n = n0 + j * 16 + ncol;
        const float bn = bias[n];
        #pragma unroll
        for (int r = 0; r < 8; ++r) {
            int g = mBase + r;
            if (g < N) hout[(size_t)g * 128 + n] = fmaxf(acc[j][r] + bn, 0.0f);
        }
    }
}

// ---------------------------------------------------------------------------
// Host orchestration
// ---------------------------------------------------------------------------
extern "C" void kernel_launch(void* const* d_in, const int* in_sizes, int n_in,
                              void* d_out, int out_size, void* d_ws, size_t ws_size,
                              hipStream_t stream) {
    const float* x   = (const float*)d_in[0];
    const int*   ei  = (const int*)d_in[1];
    const int N = in_sizes[0] / 128;
    const int E = in_sizes[1] / 2;
    const int* row = ei;          // dst
    const int* col = ei + E;      // src

    const float* Wl[3] = {(const float*)d_in[2], (const float*)d_in[5], (const float*)d_in[8]};
    const float* Wr[3] = {(const float*)d_in[3], (const float*)d_in[6], (const float*)d_in[9]};
    const float* bv[3] = {(const float*)d_in[4], (const float*)d_in[7], (const float*)d_in[10]};

    // Workspace layout
    float* agg = (float*)d_ws;                       // N*128 f32
    float* deg = agg + (size_t)N * 128;              // N f32
    unsigned short* wbase = (unsigned short*)(deg + N);
    unsigned short* WlT[3], * WrT[3];
    for (int l = 0; l < 3; ++l) {
        WlT[l] = wbase + (size_t)(2 * l)     * 128 * 128;
        WrT[l] = wbase + (size_t)(2 * l + 1) * 128 * 128;
    }

    const int TB = 256;

    // Degrees (once; identical for all layers)
    k_zero_f32<<<(N + TB - 1) / TB, TB, 0, stream>>>(deg, N);
    k_degree<<<(E + TB - 1) / TB, TB, 0, stream>>>(row, deg, E);

    // Weight conversion to transposed bf16
    for (int l = 0; l < 3; ++l) {
        k_wconv<<<(128 * 128) / TB, TB, 0, stream>>>(Wl[l], WlT[l]);
        k_wconv<<<(128 * 128) / TB, TB, 0, stream>>>(Wr[l], WrT[l]);
    }

    float* out = (float*)d_out;
    for (int l = 0; l < 3; ++l) {
        const float* hin = (l == 0) ? x : out + (size_t)(l - 1) * N * 128;
        float* hout = out + (size_t)l * N * 128;

        k_zero_f32<<<((N * 128) + TB - 1) / TB, TB, 0, stream>>>(agg, N * 128);
        k_scatter<<<((E * 32) + TB - 1) / TB, TB, 0, stream>>>(hin, row, col, agg, E);
        k_sage_gemm<<<(N + 63) / 64, TB, 0, stream>>>(agg, deg, hin,
                                                      WlT[l], WrT[l], bv[l], hout, N);
    }
}
